// BahdanauAttention_28174985462298
// MI455X (gfx1250) — matrix operand order repaired and checked by reading, not compiled, not run on hardware
//
#include <hip/hip_runtime.h>

// Bahdanau attention, fused for MI455X (gfx1250, wave32, WMMA bf16 + async LDS).
// B=32, T=2048, D=512, U=512. fp32 in/out, bf16 WMMA for the big GEMM.

#define NB 32
#define NT 2048
#define ND 512
#define NU 512

typedef __attribute__((ext_vector_type(16))) __bf16 v16bf;
typedef __attribute__((ext_vector_type(8)))  float  v8f;

union FragBF {
    uint4 u[2];
    v16bf v;
};

__device__ __forceinline__ unsigned short f2bf(float f) {
    // round-to-nearest-even fp32 -> bf16
    unsigned int u = __float_as_uint(f);
    u += 0x7FFFu + ((u >> 16) & 1u);
    return (unsigned short)(u >> 16);
}

// ---------------------------------------------------------------------------
// Kernel 0: W1 [D][U] f32  ->  W1t [U][D] bf16 (transposed, for B-fragments).
// LDS-tiled 32x32 transpose; one-time 1 MB pass.
// ---------------------------------------------------------------------------
__global__ void w1cvt_kernel(const float* __restrict__ W1,
                             unsigned short* __restrict__ W1t) {
    __shared__ float tile[32][33];
    const int u0 = blockIdx.x * 32;
    const int d0 = blockIdx.y * 32;
    const int tx = threadIdx.x;   // 32
    const int ty = threadIdx.y;   // 8
    #pragma unroll
    for (int i = 0; i < 4; ++i)
        tile[ty + i * 8][tx] = W1[(size_t)(d0 + ty + i * 8) * NU + u0 + tx];
    __syncthreads();
    #pragma unroll
    for (int i = 0; i < 4; ++i)
        W1t[(size_t)(u0 + ty + i * 8) * ND + d0 + tx] = f2bf(tile[tx][ty + i * 8]);
}

// ---------------------------------------------------------------------------
// Kernel 1: q_comb[b,u] = query[b,:] @ W2[:,u] + W2_b[u] + W1_b[u]
// ---------------------------------------------------------------------------
__global__ void qproj_kernel(const float* __restrict__ query,
                             const float* __restrict__ W2,
                             const float* __restrict__ W2b,
                             const float* __restrict__ W1b,
                             float* __restrict__ qcomb) {
    const int b = blockIdx.x;
    const int u = threadIdx.x;
    float acc = W2b[u] + W1b[u];
    const float* q = query + b * ND;
    #pragma unroll 4
    for (int d = 0; d < ND; ++d)
        acc = fmaf(q[d], W2[d * NU + u], acc);
    qcomb[b * NU + u] = acc;
}

// ---------------------------------------------------------------------------
// Kernel 2: fused score kernel.
//   score[b,t] = sum_u tanh( (values[b,t,:] . W1[:,u]) + q_comb[b,u] ) * V_w[u]
// Block = 2 waves (64 threads), 32-row T-tile per block.  values tile staged
// once as bf16; per 16-column U-slab the bf16 B-slab is staged with
// global_load_async_to_lds_b128 (ASYNCcnt path), then 16 WMMAs accumulate
// over K=512.  Per-lane tanh/V_w partial sums accumulate across slabs; one
// final __shfl_xor butterfly reduces over the 16 N-lanes.
// ---------------------------------------------------------------------------
#define MT  32          // T-rows per block
#define LDA 520         // bf16 row stride: breaks 64-bank alignment, 16B aligned
#define LDB 520

__global__ void score_kernel(const float* __restrict__ values,
                             const unsigned short* __restrict__ W1t,  // [U][D] bf16
                             const float* __restrict__ qcomb,
                             const float* __restrict__ Vw,
                             float* __restrict__ scores) {
    __shared__ unsigned short sA[MT * LDA];   // values tile, bf16 (row = t)
    __shared__ unsigned short sB[16 * LDB];   // W1t slab, bf16 (row = n)
    __shared__ float sQ[NU];
    __shared__ float sV[NU];

    const int tid  = threadIdx.x;     // 0..63
    const int b    = blockIdx.y;
    const int t0   = blockIdx.x * MT;
    const int lane = tid & 31;
    const int wave = tid >> 5;        // 0..1
    const int half = lane >> 4;       // 0..1
    const int l16  = lane & 15;       // 0..15

    // stage q_comb row + V_w (shared across all N-slabs)
    for (int i = tid; i < NU; i += 64) {
        sQ[i] = qcomb[b * NU + i];
        sV[i] = Vw[i];
    }

    // stage values[b, t0:t0+32, :] as bf16 (coalesced float4 reads)
    {
        const float4* src = reinterpret_cast<const float4*>(
            values + ((size_t)b * NT + t0) * ND);
        #pragma unroll 4
        for (int i = 0; i < 64; ++i) {
            int idx = tid + i * 64;          // 0..4095 float4s
            int row = idx >> 7;
            int c4  = idx & 127;
            float4 f = src[row * 128 + c4];
            ushort4 h;
            h.x = f2bf(f.x); h.y = f2bf(f.y); h.z = f2bf(f.z); h.w = f2bf(f.w);
            *reinterpret_cast<ushort4*>(&sA[row * LDA + c4 * 4]) = h;
        }
    }

    float scoreAcc[8];
    #pragma unroll
    for (int r = 0; r < 8; ++r) scoreAcc[r] = 0.0f;

    const unsigned short* aRow = &sA[(wave * 16 + l16) * LDA];  // A: row = M
    const unsigned short* bRow = &sB[l16 * LDB];                // B: row = N

    // per-thread base addresses for async B staging: row i gets chunk `tid`
    const unsigned sBbase = (unsigned)(size_t)(&sB[tid * 8]);

    for (int nt = 0; nt < NU / 16; ++nt) {
        const int n0 = nt * 16;
        __syncthreads();   // previous slab's WMMA reads done before overwrite

        // async-stage W1t rows n0..n0+15 -> sB (16 x b128 per thread)
        {
            const unsigned short* g = W1t + (size_t)n0 * ND + tid * 8;
            #pragma unroll
            for (int i = 0; i < 16; ++i) {
                unsigned lds = sBbase + (unsigned)(i * (LDB * 2));
                asm volatile("global_load_async_to_lds_b128 %0, %1, off"
                             :: "v"(lds), "v"(g) : "memory");
                g += ND;
            }
            asm volatile("s_wait_asynccnt 0x0" ::: "memory");
        }
        __syncthreads();

        v8f acc = {0.f, 0.f, 0.f, 0.f, 0.f, 0.f, 0.f, 0.f};
        #pragma unroll
        for (int kk = 0; kk < ND / 32; ++kk) {
            FragBF a, bb;
            // A 16x32 bf16: lane m = l16; VGPR0-3: K = half*8+0..7,
            // VGPR4-7: K = 16+half*8+0..7 -> two 16B LDS loads.
            a.u[0]  = *reinterpret_cast<const uint4*>(aRow + kk * 32 + half * 8);
            a.u[1]  = *reinterpret_cast<const uint4*>(aRow + kk * 32 + 16 + half * 8);
            // B 32x16 bf16: lane n = l16; same K layout as A.
            bb.u[0] = *reinterpret_cast<const uint4*>(bRow + kk * 32 + half * 8);
            bb.u[1] = *reinterpret_cast<const uint4*>(bRow + kk * 32 + 16 + half * 8);
            acc = __builtin_amdgcn_wmma_f32_16x16x32_bf16(
                false, a.v, false, bb.v, (short)0, acc, false, false);
        }

        // epilogue: c[r] = proj_v[M, n0+N]; N = l16, M = r + 8*half.
        // Accumulate per-lane; cross-lane N-reduction deferred to the end.
        const float qc = sQ[n0 + l16];
        const float vw = sV[n0 + l16];
        #pragma unroll
        for (int r = 0; r < 8; ++r)
            scoreAcc[r] = fmaf(tanhf(acc[r] + qc), vw, scoreAcc[r]);
    }

    // final reduction over the 16 N-lanes (xor masks stay in each 16-lane half)
    #pragma unroll
    for (int r = 0; r < 8; ++r) {
        float v = scoreAcc[r];
        v += __shfl_xor(v, 1, 32);
        v += __shfl_xor(v, 2, 32);
        v += __shfl_xor(v, 4, 32);
        v += __shfl_xor(v, 8, 32);
        if (l16 == 0)
            scores[(size_t)b * NT + t0 + wave * 16 + r + 8 * half] = v;
    }
}

// ---------------------------------------------------------------------------
// Kernel 3: in-place softmax over T per batch. One block per b, 256 threads.
// ---------------------------------------------------------------------------
__global__ void softmax_kernel(float* __restrict__ attn) {
    const int b = blockIdx.x;
    const int tid = threadIdx.x;
    float* p = attn + (size_t)b * NT;

    float vals[NT / 256];
    float mx = -INFINITY;
    #pragma unroll
    for (int i = 0; i < NT / 256; ++i) {
        vals[i] = p[tid + i * 256];
        mx = fmaxf(mx, vals[i]);
    }

    __shared__ float red[256];
    red[tid] = mx;
    __syncthreads();
    for (int s = 128; s > 0; s >>= 1) {
        if (tid < s) red[tid] = fmaxf(red[tid], red[tid + s]);
        __syncthreads();
    }
    mx = red[0];
    __syncthreads();

    float sum = 0.f;
    #pragma unroll
    for (int i = 0; i < NT / 256; ++i) {
        vals[i] = __expf(vals[i] - mx);
        sum += vals[i];
    }
    red[tid] = sum;
    __syncthreads();
    for (int s = 128; s > 0; s >>= 1) {
        if (tid < s) red[tid] += red[tid + s];
        __syncthreads();
    }
    const float inv = 1.0f / red[0];
    #pragma unroll
    for (int i = 0; i < NT / 256; ++i)
        p[tid + i * 256] = vals[i] * inv;
}

// ---------------------------------------------------------------------------
// Kernel 4: context[b,d] = sum_t attn[b,t] * values[b,t,d]
// ---------------------------------------------------------------------------
__global__ void context_kernel(const float* __restrict__ values,
                               const float* __restrict__ attn,
                               float* __restrict__ out) {
    const int b = blockIdx.x;
    const int d = threadIdx.x;
    const float* vb = values + (size_t)b * NT * ND;
    const float* ab = attn + (size_t)b * NT;
    float acc = 0.f;
    #pragma unroll 4
    for (int t = 0; t < NT; ++t)
        acc = fmaf(ab[t], vb[(size_t)t * ND + d], acc);
    out[b * ND + d] = acc;
}

// ---------------------------------------------------------------------------
// Launch.  Inputs (reference order):
//   0 values[B,T,D]  1 query[B,D]  2 W1_w[D,U]  3 W1_b[U]
//   4 W2_w[D,U]      5 W2_b[U]     6 V_w[U,1]   7 V_b[1]
// d_out: context[B,D] then attention_weights[B,T,1], fp32.
// V_b shifts all scores equally -> softmax invariant; outputs unaffected.
// ---------------------------------------------------------------------------
extern "C" void kernel_launch(void* const* d_in, const int* in_sizes, int n_in,
                              void* d_out, int out_size, void* d_ws, size_t ws_size,
                              hipStream_t stream) {
    const float* values = (const float*)d_in[0];
    const float* query  = (const float*)d_in[1];
    const float* W1w    = (const float*)d_in[2];
    const float* W1b    = (const float*)d_in[3];
    const float* W2w    = (const float*)d_in[4];
    const float* W2b    = (const float*)d_in[5];
    const float* Vw     = (const float*)d_in[6];

    float* out_ctx  = (float*)d_out;                 // [B, D]
    float* out_attn = out_ctx + NB * ND;             // [B, T]

    unsigned short* W1t = (unsigned short*)d_ws;     // [U][D] bf16, 512 KB
    float* qcomb = (float*)((char*)d_ws + (size_t)NU * ND * sizeof(unsigned short));

    w1cvt_kernel<<<dim3(NU / 32, ND / 32), dim3(32, 8), 0, stream>>>(W1w, W1t);

    qproj_kernel<<<NB, NU, 0, stream>>>(query, W2w, W2b, W1b, qcomb);

    dim3 sgrid(NT / MT, NB);                         // 64 x 32 blocks
    score_kernel<<<sgrid, 64, 0, stream>>>(values, W1t, qcomb, Vw, out_attn);

    softmax_kernel<<<NB, 256, 0, stream>>>(out_attn);

    context_kernel<<<NB, ND, 0, stream>>>(values, out_attn, out_ctx);
}
